// GAT_GCN_63402307224303
// MI455X (gfx1250) — compile-verified, matrix operand order checked by reading, not run on hardware
//
#include <hip/hip_runtime.h>

typedef __attribute__((ext_vector_type(16))) _Float16 v16h;
typedef __attribute__((ext_vector_type(8)))  float    v8f;

#define N_NODES 10000
#define N_EDGES 120000
#define E_TOT   130000
#define F_IN    66
#define HEADS   10
#define F_HEAD  66
#define GAT_OUT 660
#define GCN_OUT 1320

__device__ __forceinline__ float leakyf(float x, float s) { return x >= 0.f ? x : x * s; }

// ---------------- generic fill ----------------
__global__ void fill_f32(float* __restrict__ p, float v, long long n) {
  long long i = (long long)blockIdx.x * blockDim.x + threadIdx.x;
  if (i < n) p[i] = v;
}

// -------- pack f32 [M,K] -> f16 [M,Kp] (zero pad), optional bias+leaky --------
__global__ void pack_a(const float* __restrict__ src, const float* __restrict__ bias,
                       float slope, int M, int K, int Kp, _Float16* __restrict__ dst) {
  long long i = (long long)blockIdx.x * blockDim.x + threadIdx.x;
  long long tot = (long long)M * Kp;
  if (i >= tot) return;
  int r = (int)(i / Kp), c = (int)(i % Kp);
  float v = 0.f;
  if (c < K) {
    v = src[(long long)r * K + c];
    if (bias) v += bias[c];
    if (slope >= 0.f) v = leakyf(v, slope);
  }
  dst[i] = (_Float16)v;
}

// -------- pack weight f32 [K,N] -> transposed f16 [Np,Kp] (zero pad) --------
__global__ void pack_bt(const float* __restrict__ W, int K, int N, int Kp, int Np,
                        _Float16* __restrict__ dst) {
  long long i = (long long)blockIdx.x * blockDim.x + threadIdx.x;
  long long tot = (long long)Np * Kp;
  if (i >= tot) return;
  int n = (int)(i / Kp), k = (int)(i % Kp);
  float v = (n < N && k < K) ? W[(long long)k * N + n] : 0.f;
  dst[i] = (_Float16)v;
}

// =====================================================================
// WMMA GEMM: C[M,N] (f32, ldc) = A[M,Kp] * Bt[Np,Kp]^T, fused bias+leaky.
// Block = 256 threads = 8 waves, macro-tile 256(M) x 32(N).
// Each wave owns a 32x32 tile -> 2x2 accumulators, 4 WMMAs per K-step.
// B 32x32 K-tile is staged in LDS (2KB) once per K-step and reused by all
// 8 waves; the fill is straight-line: exactly one global_load_b64 +
// ds_store_b64 per thread per step (source pointer hoisted out of the
// loop; row = t>>3, dword pair = (t&7)*2 never crosses a B row).
// A fragments stream from global (global_load_b128) with prefetch one
// K-step ahead. Kp % 32 == 0; Bt padded to 32-multiple rows.
// =====================================================================
__global__ __launch_bounds__(256) void wmma_gemm(
    const _Float16* __restrict__ A, const _Float16* __restrict__ Bt,
    const float* __restrict__ bias, float slope,
    int M, int N, int Kp, int ldc, float* __restrict__ C) {
  __shared__ __align__(16) unsigned int lds_b[512];  // 32 cols x 32 k-halves (f16) = 2KB

  const int lane = threadIdx.x & 31;
  const int wave = threadIdx.x >> 5;
  const int lm = lane & 15, half = lane >> 4;
  const int bn0 = blockIdx.x * 32;             // N base of block
  const int bm0 = blockIdx.y * 256;            // M base of block
  const int t = threadIdx.x;

  // A row pointers for this wave's two 16-row sub-tiles (clamped: stores guarded)
  const _Float16* arow[2];
#pragma unroll
  for (int i = 0; i < 2; ++i) {
    int r = bm0 + wave * 32 + i * 16 + lm;
    if (r >= M) r = M - 1;
    arow[i] = A + (size_t)r * Kp;
  }

  // Per-thread cooperative-fill source pointer (invariant across K-steps).
  const unsigned long long* bsrc = reinterpret_cast<const unsigned long long*>(
      Bt + (size_t)(bn0 + (t >> 3)) * Kp + (t & 7) * 4);
  unsigned long long* lds64 = reinterpret_cast<unsigned long long*>(lds_b);

  v8f acc[2][2] = {{{}, {}}, {{}, {}}};

  for (int k0 = 0; k0 < Kp; k0 += 32) {
    if (k0 + 32 < Kp) {                        // prefetch next chunk (global_prefetch_b8)
      __builtin_prefetch(arow[0] + k0 + 32, 0, 1);
      __builtin_prefetch(arow[1] + k0 + 32, 0, 1);
      __builtin_prefetch(bsrc + (k0 >> 2) + 8, 0, 1);
    }

    __syncthreads();
    lds64[t] = bsrc[k0 >> 2];                  // one b64 per thread: fills 32x32 B tile
    __syncthreads();

    v16h af[2], bf[2];
#pragma unroll
    for (int i = 0; i < 2; ++i) {
      unsigned int* ap = reinterpret_cast<unsigned int*>(&af[i]);
      unsigned int* bp = reinterpret_cast<unsigned int*>(&bf[i]);
      const unsigned int* ag = reinterpret_cast<const unsigned int*>(arow[i] + k0);
      const unsigned int* lb = lds_b + (i * 16 + lm) * 16;
#pragma unroll
      for (int j = 0; j < 8; ++j) {
        // ISA 16-bit A layout: VGPR j <-> K = 16*(j/4) + 8*(lane/16) + 2*(j%4)
        int kk = ((j >> 2) << 4) + (half << 3) + ((j & 3) << 1);
        ap[j] = ag[kk >> 1];
        bp[j] = lb[kk >> 1];                   // contiguous groups of 4 -> ds_load_b128
      }
    }

#pragma unroll
    for (int i = 0; i < 2; ++i)
#pragma unroll
      for (int jn = 0; jn < 2; ++jn)
        acc[i][jn] = __builtin_amdgcn_wmma_f32_16x16x32_f16(
            false, af[i], false, bf[jn], (short)0, acc[i][jn], false, false);
  }

  // epilogue: bias + optional leaky, guarded stores
#pragma unroll
  for (int jn = 0; jn < 2; ++jn) {
    int ocol = bn0 + jn * 16 + lm;
    if (ocol >= N) continue;
    float bb = bias ? bias[ocol] : 0.f;
#pragma unroll
    for (int i = 0; i < 2; ++i) {
#pragma unroll
      for (int j = 0; j < 8; ++j) {
        int orow = bm0 + wave * 32 + i * 16 + half * 8 + j;  // C/D: VGPR j -> M=8*(lane/16)+j
        if (orow < M) {
          float v = acc[i][jn][j] + bb;
          if (slope >= 0.f) v = leakyf(v, slope);
          C[(size_t)orow * ldc + ocol] = v;
        }
      }
    }
  }
}

// ---------------- GAT attention pieces ----------------
__global__ void att_reduce(const float* __restrict__ xt, const float* __restrict__ att_s,
                           const float* __restrict__ att_d,
                           float* __restrict__ a_s, float* __restrict__ a_d) {
  int i = blockIdx.x * blockDim.x + threadIdx.x;
  if (i >= N_NODES * HEADS) return;
  int node = i / HEADS, h = i % HEADS;
  const float* xr = xt + (size_t)node * GAT_OUT + h * F_HEAD;
  const float* as = att_s + h * F_HEAD;
  const float* ad = att_d + h * F_HEAD;
  float ss = 0.f, sd = 0.f;
  for (int f = 0; f < F_HEAD; ++f) { float v = xr[f]; ss += v * as[f]; sd += v * ad[f]; }
  a_s[i] = ss; a_d[i] = sd;
}

__device__ __forceinline__ void edge_sd(const int* __restrict__ ei, int e, int& s, int& d) {
  if (e < N_EDGES) { s = ei[e]; d = ei[N_EDGES + e]; }
  else { s = e - N_EDGES; d = s; }            // self loops appended
}

__device__ __forceinline__ void atomicMaxF(float* addr, float val) {
  int old = __float_as_int(*addr);
  while (__int_as_float(old) < val) {
    int assumed = old;
    old = atomicCAS(reinterpret_cast<int*>(addr), assumed, __float_as_int(val));
    if (old == assumed) break;
  }
}

__global__ void edge_score(const int* __restrict__ ei, const float* __restrict__ a_s,
                           const float* __restrict__ a_d, float* __restrict__ esc,
                           float* __restrict__ mmax) {
  int i = blockIdx.x * blockDim.x + threadIdx.x;
  if (i >= E_TOT * HEADS) return;
  int e = i / HEADS, h = i % HEADS;
  int s, d; edge_sd(ei, e, s, d);
  float v = leakyf(a_s[s * HEADS + h] + a_d[d * HEADS + h], 0.2f);
  esc[i] = v;
  atomicMaxF(&mmax[d * HEADS + h], v);
}

__global__ void edge_exp(const int* __restrict__ ei, float* __restrict__ esc,
                         const float* __restrict__ mmax, float* __restrict__ ssum) {
  int i = blockIdx.x * blockDim.x + threadIdx.x;
  if (i >= E_TOT * HEADS) return;
  int e = i / HEADS, h = i % HEADS;
  int s, d; edge_sd(ei, e, s, d);
  float mm = mmax[d * HEADS + h];
  if (mm < -1e29f) mm = 0.f;                 // mirrors jnp.where(isfinite(m), m, 0)
  float ex = __expf(esc[i] - mm);
  esc[i] = ex;
  atomicAdd(&ssum[d * HEADS + h], ex);
}

__global__ void gat_agg(const int* __restrict__ ei, const float* __restrict__ xt,
                        const float* __restrict__ esc, const float* __restrict__ ssum,
                        float* __restrict__ h1) {
  long long i = (long long)blockIdx.x * blockDim.x + threadIdx.x;
  if (i >= (long long)E_TOT * GAT_OUT) return;
  int e = (int)(i / GAT_OUT), c = (int)(i % GAT_OUT);
  int h = c / F_HEAD;
  int s, d; edge_sd(ei, e, s, d);
  float alpha = esc[e * HEADS + h] / fmaxf(ssum[d * HEADS + h], 1e-16f);
  atomicAdd(&h1[(size_t)d * GAT_OUT + c], xt[(size_t)s * GAT_OUT + c] * alpha);
}

// ---------------- GCN pieces ----------------
__global__ void deg_count(const int* __restrict__ ei, float* __restrict__ deg) {
  int e = blockIdx.x * blockDim.x + threadIdx.x;
  if (e >= E_TOT) return;
  int s, d; edge_sd(ei, e, s, d);
  atomicAdd(&deg[d], 1.f);
}

__global__ void dinv_k(const float* __restrict__ deg, float* __restrict__ dinv) {
  int i = blockIdx.x * blockDim.x + threadIdx.x;
  if (i >= N_NODES) return;
  dinv[i] = rsqrtf(fmaxf(deg[i], 1.f));
}

__global__ void gcn_agg(const int* __restrict__ ei, const float* __restrict__ xt2,
                        const float* __restrict__ dinv, float* __restrict__ h2) {
  long long i = (long long)blockIdx.x * blockDim.x + threadIdx.x;
  if (i >= (long long)E_TOT * GCN_OUT) return;
  int e = (int)(i / GCN_OUT), c = (int)(i % GCN_OUT);
  int s, d; edge_sd(ei, e, s, d);
  float norm = dinv[s] * dinv[d];
  atomicAdd(&h2[(size_t)d * GCN_OUT + c], xt2[(size_t)s * GCN_OUT + c] * norm);
}

// ---------------- final 16 -> 1 ----------------
__global__ void final_out(const float* __restrict__ h6, const float* __restrict__ Wout,
                          const float* __restrict__ bout, float* __restrict__ out) {
  int i = blockIdx.x * blockDim.x + threadIdx.x;
  if (i >= N_NODES) return;
  float acc = bout[0];
#pragma unroll
  for (int f = 0; f < 16; ++f) acc += h6[i * 16 + f] * Wout[f];
  out[i] = acc;
}

// =====================================================================
extern "C" void kernel_launch(void* const* d_in, const int* in_sizes, int n_in,
                              void* d_out, int out_size, void* d_ws, size_t ws_size,
                              hipStream_t stream) {
  (void)in_sizes; (void)n_in; (void)out_size; (void)ws_size;
  const float* x       = (const float*)d_in[0];
  const int*   ei      = (const int*)d_in[1];
  const float* W_gat   = (const float*)d_in[2];
  const float* att_src = (const float*)d_in[3];
  const float* att_dst = (const float*)d_in[4];
  const float* b_gat   = (const float*)d_in[5];
  const float* W_gcn   = (const float*)d_in[6];
  const float* b_gcn   = (const float*)d_in[7];
  const float* W_g1    = (const float*)d_in[8];
  const float* b_g1    = (const float*)d_in[9];
  const float* W_g2    = (const float*)d_in[10];
  const float* b_g2    = (const float*)d_in[11];
  const float* W_fc1   = (const float*)d_in[12];
  const float* b_fc1   = (const float*)d_in[13];
  const float* W_fc2   = (const float*)d_in[14];
  const float* b_fc2   = (const float*)d_in[15];
  const float* W_o     = (const float*)d_in[16];
  const float* b_o     = (const float*)d_in[17];
  float* out = (float*)d_out;

  char* ws = (char*)d_ws;
  size_t off = 0;
  auto alloc = [&](size_t bytes) -> char* {
    char* p = ws + off;
    off = (off + bytes + 255) & ~(size_t)255;
    return p;
  };
  auto gsz = [](long long n, int b) -> unsigned { return (unsigned)((n + b - 1) / b); };

  // padded dims: Kp multiple of 32; Np multiple of 32 (32-wide macro-tiles)
  const int Kp1 = 96,   Np1 = 672;   // 66 -> 96, 660 -> 672
  const int Kp2 = 672,  Np2 = 1344;  // 660 -> 672, 1320 -> 1344
  const int Kp3 = 1344, Np3 = 1024;  // 1320 -> 1344, 1000 -> 1024
  const int Kp4 = 1024, Np4 = 64;
  const int Kp5 = 64,   Np5 = 32;
  const int Kp6 = 32,   Np6 = 32;    // 16 -> 32

  _Float16* Ah1 = (_Float16*)alloc((size_t)N_NODES * Kp1 * 2);
  _Float16* Bt1 = (_Float16*)alloc((size_t)Np1 * Kp1 * 2);
  float* xt1  = (float*)alloc((size_t)N_NODES * GAT_OUT * 4);
  float* a_s  = (float*)alloc((size_t)N_NODES * HEADS * 4);
  float* a_d  = (float*)alloc((size_t)N_NODES * HEADS * 4);
  float* mmax = (float*)alloc((size_t)N_NODES * HEADS * 4);
  float* ssum = (float*)alloc((size_t)N_NODES * HEADS * 4);
  float* esc  = (float*)alloc((size_t)E_TOT * HEADS * 4);
  float* h1   = (float*)alloc((size_t)N_NODES * GAT_OUT * 4);
  _Float16* Ah2 = (_Float16*)alloc((size_t)N_NODES * Kp2 * 2);
  _Float16* Bt2 = (_Float16*)alloc((size_t)Np2 * Kp2 * 2);
  float* xt2  = (float*)alloc((size_t)N_NODES * GCN_OUT * 4);
  float* deg  = (float*)alloc((size_t)N_NODES * 4);
  float* dinv = (float*)alloc((size_t)N_NODES * 4);
  float* h2   = (float*)alloc((size_t)N_NODES * GCN_OUT * 4);
  _Float16* Ah3 = (_Float16*)alloc((size_t)N_NODES * Kp3 * 2);
  _Float16* Bt3 = (_Float16*)alloc((size_t)Np3 * Kp3 * 2);
  float* h3   = (float*)alloc((size_t)N_NODES * 1000 * 4);
  _Float16* Ah4 = (_Float16*)alloc((size_t)N_NODES * Kp4 * 2);
  _Float16* Bt4 = (_Float16*)alloc((size_t)Np4 * Kp4 * 2);
  float* h4   = (float*)alloc((size_t)N_NODES * 64 * 4);
  _Float16* Ah5 = (_Float16*)alloc((size_t)N_NODES * Kp5 * 2);
  _Float16* Bt5 = (_Float16*)alloc((size_t)Np5 * Kp5 * 2);
  float* h5   = (float*)alloc((size_t)N_NODES * 32 * 4);
  _Float16* Ah6 = (_Float16*)alloc((size_t)N_NODES * Kp6 * 2);
  _Float16* Bt6 = (_Float16*)alloc((size_t)Np6 * Kp6 * 2);
  float* h6   = (float*)alloc((size_t)N_NODES * 16 * 4);

  const int B = 256;
  auto launch_gemm = [&](const _Float16* A, const _Float16* Bt, const float* bias,
                         float slope, int M, int N, int Kp, int ldc, float* C) {
    dim3 grid((unsigned)((N + 31) / 32), (unsigned)((M + 255) / 256));
    wmma_gemm<<<grid, B, 0, stream>>>(A, Bt, bias, slope, M, N, Kp, ldc, C);
  };

  // --- GAT linear: xt1 = x @ W_gat ---
  pack_a<<<gsz((long long)N_NODES * Kp1, B), B, 0, stream>>>(x, nullptr, -1.f, N_NODES, F_IN, Kp1, Ah1);
  pack_bt<<<gsz((long long)Np1 * Kp1, B), B, 0, stream>>>(W_gat, F_IN, GAT_OUT, Kp1, Np1, Bt1);
  launch_gemm(Ah1, Bt1, nullptr, -1.f, N_NODES, GAT_OUT, Kp1, GAT_OUT, xt1);

  // --- GAT attention / segment softmax ---
  att_reduce<<<gsz((long long)N_NODES * HEADS, B), B, 0, stream>>>(xt1, att_src, att_dst, a_s, a_d);
  fill_f32<<<gsz((long long)N_NODES * HEADS, B), B, 0, stream>>>(mmax, -1e30f, (long long)N_NODES * HEADS);
  fill_f32<<<gsz((long long)N_NODES * HEADS, B), B, 0, stream>>>(ssum, 0.f, (long long)N_NODES * HEADS);
  fill_f32<<<gsz((long long)N_NODES * GAT_OUT, B), B, 0, stream>>>(h1, 0.f, (long long)N_NODES * GAT_OUT);
  edge_score<<<gsz((long long)E_TOT * HEADS, B), B, 0, stream>>>(ei, a_s, a_d, esc, mmax);
  edge_exp<<<gsz((long long)E_TOT * HEADS, B), B, 0, stream>>>(ei, esc, mmax, ssum);
  gat_agg<<<gsz((long long)E_TOT * GAT_OUT, B), B, 0, stream>>>(ei, xt1, esc, ssum, h1);

  // --- GCN linear: xt2 = leaky(h1 + b_gat) @ W_gcn ---
  pack_a<<<gsz((long long)N_NODES * Kp2, B), B, 0, stream>>>(h1, b_gat, 0.01f, N_NODES, GAT_OUT, Kp2, Ah2);
  pack_bt<<<gsz((long long)Np2 * Kp2, B), B, 0, stream>>>(W_gcn, GAT_OUT, GCN_OUT, Kp2, Np2, Bt2);
  launch_gemm(Ah2, Bt2, nullptr, -1.f, N_NODES, GCN_OUT, Kp2, GCN_OUT, xt2);

  // --- GCN normalize + aggregate ---
  fill_f32<<<gsz(N_NODES, B), B, 0, stream>>>(deg, 0.f, N_NODES);
  deg_count<<<gsz(E_TOT, B), B, 0, stream>>>(ei, deg);
  dinv_k<<<gsz(N_NODES, B), B, 0, stream>>>(deg, dinv);
  fill_f32<<<gsz((long long)N_NODES * GCN_OUT, B), B, 0, stream>>>(h2, 0.f, (long long)N_NODES * GCN_OUT);
  gcn_agg<<<gsz((long long)E_TOT * GCN_OUT, B), B, 0, stream>>>(ei, xt2, dinv, h2);

  // --- dense tail ---
  pack_a<<<gsz((long long)N_NODES * Kp3, B), B, 0, stream>>>(h2, b_gcn, 0.01f, N_NODES, GCN_OUT, Kp3, Ah3);
  pack_bt<<<gsz((long long)Np3 * Kp3, B), B, 0, stream>>>(W_g1, GCN_OUT, 1000, Kp3, Np3, Bt3);
  launch_gemm(Ah3, Bt3, b_g1, 0.01f, N_NODES, 1000, Kp3, 1000, h3);

  pack_a<<<gsz((long long)N_NODES * Kp4, B), B, 0, stream>>>(h3, nullptr, -1.f, N_NODES, 1000, Kp4, Ah4);
  pack_bt<<<gsz((long long)Np4 * Kp4, B), B, 0, stream>>>(W_g2, 1000, 64, Kp4, Np4, Bt4);
  launch_gemm(Ah4, Bt4, b_g2, 0.01f, N_NODES, 64, Kp4, 64, h4);

  pack_a<<<gsz((long long)N_NODES * Kp5, B), B, 0, stream>>>(h4, nullptr, -1.f, N_NODES, 64, Kp5, Ah5);
  pack_bt<<<gsz((long long)Np5 * Kp5, B), B, 0, stream>>>(W_fc1, 64, 32, Kp5, Np5, Bt5);
  launch_gemm(Ah5, Bt5, b_fc1, 0.01f, N_NODES, 32, Kp5, 32, h5);

  pack_a<<<gsz((long long)N_NODES * Kp6, B), B, 0, stream>>>(h5, nullptr, -1.f, N_NODES, 32, Kp6, Ah6);
  pack_bt<<<gsz((long long)Np6 * Kp6, B), B, 0, stream>>>(W_fc2, 32, 16, Kp6, Np6, Bt6);
  launch_gemm(Ah6, Bt6, b_fc2, 0.01f, N_NODES, 16, Kp6, 16, h6);

  final_out<<<gsz(N_NODES, B), B, 0, stream>>>(h6, W_o, b_o, out);
}
